// SuperGenerator_80942953660978
// MI455X (gfx1250) — compile-verified, matrix operand order checked by reading, not run on hardware
//
// MI455X (gfx1250) implementation of the Mamba-UNet generator.
// NHWC activations; every conv/deconv/projection lowered to an f16-operand /
// f32-accumulate WMMA GEMM (v_wmma_f32_16x16x32_f16). Each wave computes a
// 32x32 macro-tile (2x2 WMMA tiles) so each A/B fragment load feeds two
// WMMAs (2 loads/WMMA). im2col / im2colT gather kernels feed the GEMM; fp32
// epilogues fuse bias + (lrelu|relu|softplus) + residual-add. InstanceNorm /
// LayerNorm / SSM scan / channel-attention are fp32 kernels; the SSM scan is
// fused with the D-skip and SiLU(z) gate and emits the out_proj A-operand in
// f16 directly. Workspace arena requirement ~262 MiB.

#include <hip/hip_runtime.h>

typedef __attribute__((ext_vector_type(16))) _Float16 v16h;
typedef __attribute__((ext_vector_type(8)))  _Float16 v8h;
typedef __attribute__((ext_vector_type(8)))  float    v8f;

#define EPSN 1e-5f

// ============================ WMMA GEMM ====================================
// C[M][Nstride] = act(A[M][Kp] * Bw[Np][Kp]^T + bias) (+ addsrc)
// A, Bw f16 row-major over K (Kp multiple of 32; Bw padded to 32 N-rows).
// One wave computes a 32x32 macro-tile (2x2 of 16x16 WMMA tiles).
__global__ __launch_bounds__(256)
void gemm_f16_wmma(const _Float16* __restrict__ A, const _Float16* __restrict__ Bw,
                   const float* __restrict__ bias, const float* __restrict__ addsrc,
                   float* __restrict__ C, int M, int Nvalid, int Kp, int Nstride, int act)
{
  const int wave   = threadIdx.x >> 5;
  const int lane   = threadIdx.x & 31;
  const int tilesN = (Nvalid + 31) >> 5;
  const int tilesM = (M + 31) >> 5;
  const int gid    = blockIdx.x * 8 + wave;
  if (gid >= tilesM * tilesN) return;            // wave-uniform: EXEC all-1s below
  const int tm   = gid / tilesN;
  const int tn   = gid - tm * tilesN;
  const int row  = lane & 15;                    // M-row (A) / N-row (B)
  const int koff = (lane >> 4) << 3;             // K half: 0 or 8

  const _Float16* aRow0 = A  + (size_t)(tm * 32 + row) * Kp + koff;
  const _Float16* aRow1 = aRow0 + (size_t)16 * Kp;
  const _Float16* bRow0 = Bw + (size_t)(tn * 32 + row) * Kp + koff;
  const _Float16* bRow1 = bRow0 + (size_t)16 * Kp;

  v8f acc00 = {}, acc01 = {}, acc10 = {}, acc11 = {};
  for (int k = 0; k < Kp; k += 32) {
    v8h a00 = *(const v8h*)(aRow0 + k);
    v8h a01 = *(const v8h*)(aRow0 + k + 16);
    v8h a10 = *(const v8h*)(aRow1 + k);
    v8h a11 = *(const v8h*)(aRow1 + k + 16);
    v8h b00 = *(const v8h*)(bRow0 + k);
    v8h b01 = *(const v8h*)(bRow0 + k + 16);
    v8h b10 = *(const v8h*)(bRow1 + k);
    v8h b11 = *(const v8h*)(bRow1 + k + 16);
    v16h a0, a1, b0, b1;
#pragma unroll
    for (int i = 0; i < 8; ++i) {
      a0[i] = a00[i]; a0[i + 8] = a01[i];
      a1[i] = a10[i]; a1[i + 8] = a11[i];
      b0[i] = b00[i]; b0[i + 8] = b01[i];
      b1[i] = b10[i]; b1[i + 8] = b11[i];
    }
    acc00 = __builtin_amdgcn_wmma_f32_16x16x32_f16(false, a0, false, b0, (short)0, acc00, false, false);
    acc01 = __builtin_amdgcn_wmma_f32_16x16x32_f16(false, a0, false, b1, (short)0, acc01, false, false);
    acc10 = __builtin_amdgcn_wmma_f32_16x16x32_f16(false, a1, false, b0, (short)0, acc10, false, false);
    acc11 = __builtin_amdgcn_wmma_f32_16x16x32_f16(false, a1, false, b1, (short)0, acc11, false, false);
  }

  // C/D layout: lane&15 = N-col, (lane>>4)*8 + r = M-row within 16x16 tile.
  const int ncol  = lane & 15;
  const int rbase = (lane >> 4) << 3;
#pragma unroll
  for (int i = 0; i < 2; ++i) {
#pragma unroll
    for (int j = 0; j < 2; ++j) {
      const v8f* accp = (i == 0) ? (j == 0 ? &acc00 : &acc01)
                                 : (j == 0 ? &acc10 : &acc11);
      int n = tn * 32 + j * 16 + ncol;
      if (n >= Nvalid) continue;
      float bn = bias ? bias[n] : 0.f;
#pragma unroll
      for (int r = 0; r < 8; ++r) {
        int m = tm * 32 + i * 16 + rbase + r;
        if (m >= M) continue;
        float v = (*accp)[r] + bn;
        if (act == 1)      v = fmaxf(v, 0.f);                      // relu
        else if (act == 2) v = (v >= 0.f) ? v : 0.2f * v;          // lrelu(0.2)
        else if (act == 3) v = (v > 20.f) ? v : log1pf(__expf(v)); // softplus
        if (addsrc) v += addsrc[(size_t)m * Nstride + n];
        C[(size_t)m * Nstride + n] = v;
      }
    }
  }
}

// ======================= weight packing (fp32 -> f16) ======================
// Regular conv / linear weights, already [N][K] contiguous. N padded to 32.
__global__ void pack_w_f16(const float* __restrict__ w, _Float16* __restrict__ out,
                           int Nv, int K, int Kp)
{
  int Np  = (Nv + 31) & ~31;
  size_t idx = (size_t)blockIdx.x * blockDim.x + threadIdx.x;
  if (idx >= (size_t)Np * Kp) return;
  int k = (int)(idx % Kp), n = (int)(idx / Kp);
  float v = (n < Nv && k < K) ? w[(size_t)n * K + k] : 0.f;
  out[idx] = (_Float16)v;
}

// ConvTranspose weights (IC, OC, KH, KW) -> out[oc][ic*KK + tap]; OC pad to 32.
__global__ void pack_wT_f16(const float* __restrict__ w, _Float16* __restrict__ out,
                            int IC, int OC, int KK, int Kp)
{
  int Np  = (OC + 31) & ~31;
  size_t idx = (size_t)blockIdx.x * blockDim.x + threadIdx.x;
  if (idx >= (size_t)Np * Kp) return;
  int k = (int)(idx % Kp), oc = (int)(idx / Kp);
  float v = 0.f;
  if (oc < OC && k < IC * KK) {
    int t = k % KK, ic = k / KK;
    v = w[((size_t)ic * OC + oc) * KK + t];
  }
  out[idx] = (_Float16)v;
}

// ============================== im2col =====================================
// NHWC fp32 input -> f16 col[M][Kp], k = (ic*KH + ky)*KW + kx
__global__ void im2col_f16(const float* __restrict__ in, _Float16* __restrict__ col,
                           int N, int Hin, int Win, int Cin, int KH, int KW,
                           int stride, int pad, int Hout, int Wout, int Kp)
{
  size_t idx   = (size_t)blockIdx.x * blockDim.x + threadIdx.x;
  size_t total = (size_t)N * Hout * Wout * Kp;
  if (idx >= total) return;
  int k    = (int)(idx % Kp);
  size_t t = idx / Kp;
  float v = 0.f;
  if (k < Cin * KH * KW) {
    int kx = k % KW, ky = (k / KW) % KH, ic = k / (KW * KH);
    int ox = (int)(t % Wout);
    int oy = (int)((t / Wout) % Hout);
    int n  = (int)(t / ((size_t)Wout * Hout));
    int iy = oy * stride - pad + ky;
    int ix = ox * stride - pad + kx;
    if (iy >= 0 && iy < Hin && ix >= 0 && ix < Win)
      v = in[(((size_t)n * Hin + iy) * Win + ix) * Cin + ic];
  }
  col[idx] = (_Float16)v;
}

// Transposed-conv gather (k=4, s=2, p=1): y[oy,ox] uses x[(oy+1-ky)/2, (ox+1-kx)/2]
__global__ void im2colT_f16(const float* __restrict__ in, _Float16* __restrict__ col,
                            int N, int Hin, int Win, int Cin, int Hout, int Wout, int Kp)
{
  size_t idx   = (size_t)blockIdx.x * blockDim.x + threadIdx.x;
  size_t total = (size_t)N * Hout * Wout * Kp;
  if (idx >= total) return;
  int k    = (int)(idx % Kp);
  size_t t = idx / Kp;
  float v = 0.f;
  if (k < Cin * 16) {
    int tap = k & 15, kx = tap & 3, ky = tap >> 2, ic = k >> 4;
    int ox = (int)(t % Wout);
    int oy = (int)((t / Wout) % Hout);
    int n  = (int)(t / ((size_t)Wout * Hout));
    int ny = oy + 1 - ky, nx = ox + 1 - kx;
    if (ny >= 0 && nx >= 0 && !(ny & 1) && !(nx & 1)) {
      int iy = ny >> 1, ix = nx >> 1;
      if (iy < Hin && ix < Win)
        v = in[(((size_t)n * Hin + iy) * Win + ix) * Cin + ic];
    }
  }
  col[idx] = (_Float16)v;
}

// ========================= layout / elementwise ============================
__global__ void nchw_to_nhwc(const float* __restrict__ in, float* __restrict__ out,
                             int N, int C, int H, int W)
{
  size_t idx = (size_t)blockIdx.x * blockDim.x + threadIdx.x;
  if (idx >= (size_t)N * C * H * W) return;
  int x = (int)(idx % W);
  int y = (int)((idx / W) % H);
  int c = (int)((idx / ((size_t)W * H)) % C);
  int n = (int)(idx / ((size_t)W * H * C));
  out[(((size_t)n * H + y) * W + x) * C + c] = in[idx];
}

__global__ void add_f32(const float* __restrict__ a, const float* __restrict__ b,
                        float* __restrict__ c, size_t n)
{
  size_t i = (size_t)blockIdx.x * blockDim.x + threadIdx.x;
  if (i < n) c[i] = a[i] + b[i];
}

// InstanceNorm (affine=False), NHWC, in-place, optional activation.
// One block per (n, c); population variance.
__global__ __launch_bounds__(256)
void inorm_nhwc(float* __restrict__ x, int C, int S, int act)
{
  int n = blockIdx.x / C, c = blockIdx.x % C;
  float* base = x + (size_t)n * S * C + c;
  float s1 = 0.f, s2 = 0.f;
  for (int i = threadIdx.x; i < S; i += 256) {
    float v = base[(size_t)i * C]; s1 += v; s2 += v * v;
  }
  __shared__ float r1[256], r2[256];
  r1[threadIdx.x] = s1; r2[threadIdx.x] = s2; __syncthreads();
  for (int o = 128; o > 0; o >>= 1) {
    if (threadIdx.x < o) { r1[threadIdx.x] += r1[threadIdx.x + o]; r2[threadIdx.x] += r2[threadIdx.x + o]; }
    __syncthreads();
  }
  float mean = r1[0] / S;
  float var  = r2[0] / S - mean * mean;
  float inv  = rsqrtf(var + EPSN);
  for (int i = threadIdx.x; i < S; i += 256) {
    float v = (base[(size_t)i * C] - mean) * inv;
    if (act == 1)      v = fmaxf(v, 0.f);
    else if (act == 2) v = (v >= 0.f) ? v : 0.2f * v;
    base[(size_t)i * C] = v;
  }
}

// =============================== Mamba =====================================
// Token LayerNorm (C==blockDim==128), writes f16 GEMM A-operand.
__global__ __launch_bounds__(128)
void ln_tokens_f16(const float* __restrict__ x, const float* __restrict__ g,
                   const float* __restrict__ b, _Float16* __restrict__ out, int C)
{
  int t = blockIdx.x, c = threadIdx.x;
  __shared__ float red[128];
  float v = x[(size_t)t * C + c];
  red[c] = v; __syncthreads();
  for (int o = 64; o > 0; o >>= 1) { if (c < o) red[c] += red[c + o]; __syncthreads(); }
  float mean = red[0] / C;
  __syncthreads();
  float d = v - mean;
  red[c] = d * d; __syncthreads();
  for (int o = 64; o > 0; o >>= 1) { if (c < o) red[c] += red[c + o]; __syncthreads(); }
  float inv = rsqrtf(red[0] / C + EPSN);
  out[(size_t)t * C + c] = (_Float16)(d * inv * g[c] + b[c]);
}

// Causal depthwise conv1d (width 4) + SiLU. xz=[M][2*di] (xp = cols [0,di)).
// Writes fp32 xc and its f16 copy (x_proj A-operand).
__global__ void dwconv_silu(const float* __restrict__ xz, const float* __restrict__ w,
                            const float* __restrict__ b, float* __restrict__ xc,
                            _Float16* __restrict__ xch, int B, int L, int di)
{
  size_t idx = (size_t)blockIdx.x * blockDim.x + threadIdx.x;
  if (idx >= (size_t)B * L * di) return;
  int c    = (int)(idx % di);
  size_t t = idx / di;
  int l    = (int)(t % L);
  int n    = (int)(t / L);
  float acc = b[c];
#pragma unroll
  for (int j = 0; j < 4; ++j) {
    int ll = l - 3 + j;
    if (ll >= 0) acc += w[c * 4 + j] * xz[((size_t)n * L + ll) * (2 * di) + c];
  }
  float s = acc / (1.f + __expf(-acc));          // silu
  xc[idx]  = s;
  xch[idx] = (_Float16)s;
}

// Gather dt columns (dbc[:, :dr]) into zero-padded f16 [M][32].
__global__ void pack_dtcols_f16(const float* __restrict__ dbc, _Float16* __restrict__ out,
                                int M, int dr, int dbcStride)
{
  size_t idx = (size_t)blockIdx.x * blockDim.x + threadIdx.x;
  if (idx >= (size_t)M * 32) return;
  int k    = (int)(idx % 32);
  size_t t = idx / 32;
  float v = (k < dr) ? dbc[t * dbcStride + k] : 0.f;
  out[idx] = (_Float16)v;
}

// Selective-scan: h_s = exp(dt*A_s)*h_s + dt*x*B_s ; y = sum_s h_s*C_s,
// fused with y += x*D and SiLU(z) gate. One block per batch, thread per channel.
__global__ __launch_bounds__(256)
void ssm_scan(const float* __restrict__ dt, const float* __restrict__ xc,
              const float* __restrict__ dbc, const float* __restrict__ xz,
              const float* __restrict__ Alog, const float* __restrict__ Dp,
              _Float16* __restrict__ yh, int L)
{
  const int di = 256, ds = 16, dr = 8, dbcS = 48;
  int n = blockIdx.x, c = threadIdx.x;
  float a[16], h[16];
#pragma unroll
  for (int s = 0; s < 16; ++s) { a[s] = -__expf(Alog[c * ds + s]); h[s] = 0.f; }
  float Dc = Dp[c];
  __shared__ float shB[16], shC[16];
  for (int l = 0; l < L; ++l) {
    size_t m = (size_t)n * L + l;
    if (threadIdx.x < 16)       shB[threadIdx.x]      = dbc[m * dbcS + dr + threadIdx.x];
    else if (threadIdx.x < 32)  shC[threadIdx.x - 16] = dbc[m * dbcS + dr + ds + (threadIdx.x - 16)];
    __syncthreads();
    float dtv = dt[m * di + c];
    float xv  = xc[m * di + c];
    float y = 0.f;
#pragma unroll
    for (int s = 0; s < 16; ++s) {
      h[s] = __expf(dtv * a[s]) * h[s] + dtv * xv * shB[s];
      y += h[s] * shC[s];
    }
    float zv  = xz[m * (2 * di) + di + c];
    float out = (y + xv * Dc) * (zv / (1.f + __expf(-zv)));
    yh[m * di + c] = (_Float16)out;
    __syncthreads();
  }
}

// ========================= channel attention ===============================
__global__ __launch_bounds__(128)
void ca_reduce(const float* __restrict__ x, float* __restrict__ s, int L, int C)
{
  int n = blockIdx.x, c = threadIdx.x;
  const float* base = x + (size_t)n * L * C + c;
  float acc = 0.f;
  for (int i = 0; i < L; ++i) acc += base[(size_t)i * C];
  s[n * C + c] = acc / (float)L;
}

__global__ __launch_bounds__(128)
void ca_mlp(const float* __restrict__ s, const float* __restrict__ w1, const float* __restrict__ b1,
            const float* __restrict__ w2, const float* __restrict__ b2,
            float* __restrict__ a, int C, int Hd)
{
  int n = blockIdx.x, c = threadIdx.x;
  __shared__ float sh[128], hh[8];
  sh[c] = s[n * C + c]; __syncthreads();
  if (c < Hd) {
    float acc = b1[c];
    for (int j = 0; j < C; ++j) acc += w1[c * C + j] * sh[j];
    hh[c] = fmaxf(acc, 0.f);
  }
  __syncthreads();
  float acc = b2[c];
  for (int j = 0; j < Hd; ++j) acc += w2[c * Hd + j] * hh[j];
  a[n * C + c] = 1.f / (1.f + __expf(-acc));
}

__global__ void ca_scale(const float* __restrict__ x, const float* __restrict__ a,
                         float* __restrict__ y, int L, int C)
{
  size_t idx = (size_t)blockIdx.x * blockDim.x + threadIdx.x;
  if (idx >= (size_t)2 * L * C) return;
  int c = (int)(idx % C);
  int n = (int)(idx / ((size_t)L * C));
  y[idx] = x[idx] * a[n * C + c];
}

// =============================== output ====================================
__global__ void final_tanh_nchw(const float* __restrict__ t6, float* __restrict__ out,
                                int N, int H, int W, int Np)
{
  size_t idx = (size_t)blockIdx.x * blockDim.x + threadIdx.x;
  if (idx >= (size_t)N * 3 * H * W) return;
  int x = (int)(idx % W);
  int y = (int)((idx / W) % H);
  int c = (int)((idx / ((size_t)W * H)) % 3);
  int n = (int)(idx / ((size_t)W * H * 3));
  out[idx] = tanhf(t6[(((size_t)n * H + y) * W + x) * Np + c]);
}

// =============================== host ======================================
static inline unsigned gcount(size_t total, int tpb) { return (unsigned)((total + tpb - 1) / tpb); }

extern "C" void kernel_launch(void* const* d_in, const int* in_sizes, int n_in,
                              void* d_out, int out_size, void* d_ws, size_t ws_size,
                              hipStream_t stream)
{
  (void)in_sizes; (void)n_in; (void)out_size; (void)ws_size;
  auto P = [&](int i) { return (const float*)d_in[i]; };

  // ---- workspace arena (~262 MiB) ----
  char* ws = (char*)d_ws;
  size_t off = 0;
  auto arena = [&](size_t bytes) -> char* {
    char* p = ws + off;
    off += (bytes + 255) & ~(size_t)255;
    return p;
  };
  float*    x_nhwc = (float*)arena(32768ull * 6 * 4);
  _Float16* col    = (_Float16*)arena(134217728ull);           // 128 MiB (d2 im2colT)
  _Float16* wpack  = (_Float16*)arena(2ull << 20);
  float*    e1keep = (float*)arena(8192ull * 64 * 4);
  float*    actA   = (float*)arena(33554432ull);
  float*    actB   = (float*)arena(33554432ull);
  float*    actC   = (float*)arena(33554432ull);
  float*    xzb    = (float*)arena(8192ull * 512 * 4);
  float*    xcb    = (float*)arena(8192ull * 256 * 4);
  float*    dtbuf  = (float*)arena(8192ull * 256 * 4);
  float*    dbcb   = (float*)arena(8192ull * 48 * 4);
  float*    sbuf   = (float*)arena(2 * 128 * 4);
  float*    abuf   = (float*)arena(2 * 128 * 4);
  // f16 Mamba operand slices carved out of `col` (unused by convs during Mamba)
  _Float16* lnh = col;
  _Float16* xch = (_Float16*)((char*)col + (8ull  << 20));
  _Float16* dth = (_Float16*)((char*)col + (16ull << 20));
  _Float16* yh  = (_Float16*)((char*)col + (20ull << 20));

  auto gemm = [&](const _Float16* A, const float* bias, const float* add, float* C,
                  int M, int Nv, int Kp, int Nstride, int act) {
    int tiles = ((M + 31) / 32) * ((Nv + 31) / 32);
    gemm_f16_wmma<<<(tiles + 7) / 8, 256, 0, stream>>>(A, wpack, bias, add, C, M, Nv, Kp, Nstride, act);
  };
  auto pack = [&](const float* w, int Nv, int K, int Kp) {
    size_t total = (size_t)((Nv + 31) & ~31) * Kp;
    pack_w_f16<<<gcount(total, 256), 256, 0, stream>>>(w, wpack, Nv, K, Kp);
  };
  auto packT = [&](const float* w, int IC, int OC, int Kp) {
    size_t total = (size_t)((OC + 31) & ~31) * Kp;
    pack_wT_f16<<<gcount(total, 256), 256, 0, stream>>>(w, wpack, IC, OC, 16, Kp);
  };
  auto im2col = [&](const float* in, int N, int H, int W, int C, int KH, int KW,
                    int s, int p, int Ho, int Wo, int Kp) {
    size_t total = (size_t)N * Ho * Wo * Kp;
    im2col_f16<<<gcount(total, 256), 256, 0, stream>>>(in, col, N, H, W, C, KH, KW, s, p, Ho, Wo, Kp);
  };
  auto im2colT = [&](const float* in, int N, int H, int W, int C, int Ho, int Wo, int Kp) {
    size_t total = (size_t)N * Ho * Wo * Kp;
    im2colT_f16<<<gcount(total, 256), 256, 0, stream>>>(in, col, N, H, W, C, Ho, Wo, Kp);
  };
  auto inorm = [&](float* x, int N, int C, int S, int act) {
    inorm_nhwc<<<N * C, 256, 0, stream>>>(x, C, S, act);
  };

  // Mamba block: d=128, di=256, dr=8, ds=16. in/out NHWC-token fp32 [Bn*L][128].
  auto mamba = [&](const float* in, float* out, int base, int Bn, int L) {
    int M = Bn * L;
    const float *ln_g = P(base + 0), *ln_b = P(base + 1), *in_w = P(base + 2),
                *c1w = P(base + 3), *c1b = P(base + 4), *xw = P(base + 5),
                *dtw = P(base + 6), *dtbias = P(base + 7), *Alog = P(base + 8),
                *Dp = P(base + 9), *ow = P(base + 10);
    ln_tokens_f16<<<M, 128, 0, stream>>>(in, ln_g, ln_b, lnh, 128);
    pack(in_w, 512, 128, 128);
    gemm(lnh, nullptr, nullptr, xzb, M, 512, 128, 512, 0);       // in_proj -> xz
    dwconv_silu<<<gcount((size_t)M * 256, 256), 256, 0, stream>>>(xzb, c1w, c1b, xcb, xch, Bn, L, 256);
    pack(xw, 40, 256, 256);
    gemm(xch, nullptr, nullptr, dbcb, M, 40, 256, 48, 0);        // x_proj -> [dt|B|C]
    pack_dtcols_f16<<<gcount((size_t)M * 32, 256), 256, 0, stream>>>(dbcb, dth, M, 8, 48);
    pack(dtw, 256, 8, 32);
    gemm(dth, dtbias, nullptr, dtbuf, M, 256, 32, 256, 3);       // dt_proj + softplus
    ssm_scan<<<Bn, 256, 0, stream>>>(dtbuf, xcb, dbcb, xzb, Alog, Dp, yh, L);
    pack(ow, 128, 256, 256);
    gemm(yh, nullptr, in, out, M, 128, 256, 128, 0);             // out_proj + residual
  };

  // ------------------------------ pipeline ---------------------------------
  nchw_to_nhwc<<<gcount(2ull * 6 * 128 * 128, 256), 256, 0, stream>>>((const float*)d_in[0], x_nhwc, 2, 6, 128, 128);

  // e1: conv 6->64 k4 s2 p1 + lrelu
  im2col(x_nhwc, 2, 128, 128, 6, 4, 4, 2, 1, 64, 64, 96);
  pack(P(1), 64, 96, 96);
  gemm(col, P(2), nullptr, actA, 8192, 64, 96, 64, 2);
  // rb1
  im2col(actA, 2, 64, 64, 64, 3, 3, 1, 1, 64, 64, 576);
  pack(P(3), 64, 576, 576);
  gemm(col, P(4), nullptr, actB, 8192, 64, 576, 64, 0);
  inorm(actB, 2, 64, 4096, 1);
  im2col(actB, 2, 64, 64, 64, 3, 3, 1, 1, 64, 64, 576);
  pack(P(5), 64, 576, 576);
  gemm(col, P(6), nullptr, actC, 8192, 64, 576, 64, 0);
  inorm(actC, 2, 64, 4096, 0);
  add_f32<<<gcount(8192ull * 64, 256), 256, 0, stream>>>(actA, actC, e1keep, 8192ull * 64);

  // e2: conv 64->128 k4 s2 p1, inorm, lrelu
  im2col(e1keep, 2, 64, 64, 64, 4, 4, 2, 1, 32, 32, 1024);
  pack(P(7), 128, 1024, 1024);
  gemm(col, P(8), nullptr, actA, 2048, 128, 1024, 128, 0);
  inorm(actA, 2, 128, 1024, 2);

  mamba(actA, actB, 9, 2, 1024);    // m1
  mamba(actB, actA, 20, 2, 1024);   // m2

  // channel attention
  ca_reduce<<<2, 128, 0, stream>>>(actA, sbuf, 1024, 128);
  ca_mlp<<<2, 128, 0, stream>>>(sbuf, P(31), P(32), P(33), P(34), abuf, 128, 8);
  ca_scale<<<gcount(2048ull * 128, 256), 256, 0, stream>>>(actA, abuf, actC, 1024, 128);

  // bottleneck conv 128->256 k3 p1
  im2col(actC, 2, 32, 32, 128, 3, 3, 1, 1, 32, 32, 1152);
  pack(P(35), 256, 1152, 1152);
  gemm(col, P(36), nullptr, actA, 2048, 256, 1152, 256, 0);

  // d1: convT 256->128, inorm, relu, mamba m3
  im2colT(actA, 2, 32, 32, 256, 64, 64, 4096);
  packT(P(37), 256, 128, 4096);
  gemm(col, P(38), nullptr, actB, 8192, 128, 4096, 128, 0);
  inorm(actB, 2, 128, 4096, 1);
  mamba(actB, actA, 39, 2, 4096);   // m3

  // skip: 1x1 conv 64->128 on e1, added to m3 output
  im2col(e1keep, 2, 64, 64, 64, 1, 1, 1, 0, 64, 64, 64);
  pack(P(50), 128, 64, 64);
  gemm(col, P(51), actA, actC, 8192, 128, 64, 128, 0);

  // d2: convT 128->64, inorm, relu
  im2colT(actC, 2, 64, 64, 128, 128, 128, 2048);
  packT(P(52), 128, 64, 2048);
  gemm(col, P(53), nullptr, actA, 32768, 64, 2048, 64, 0);
  inorm(actA, 2, 64, 16384, 1);
  // rb2
  im2col(actA, 2, 128, 128, 64, 3, 3, 1, 1, 128, 128, 576);
  pack(P(54), 64, 576, 576);
  gemm(col, P(55), nullptr, actB, 32768, 64, 576, 64, 0);
  inorm(actB, 2, 64, 16384, 1);
  im2col(actB, 2, 128, 128, 64, 3, 3, 1, 1, 128, 128, 576);
  pack(P(56), 64, 576, 576);
  gemm(col, P(57), nullptr, actC, 32768, 64, 576, 64, 0);
  inorm(actC, 2, 64, 16384, 0);
  add_f32<<<gcount(32768ull * 64, 256), 256, 0, stream>>>(actA, actC, actB, 32768ull * 64);

  // fin: conv 64->3 k3 p1 (N padded in pack), then tanh + NHWC->NCHW
  im2col(actB, 2, 128, 128, 64, 3, 3, 1, 1, 128, 128, 576);
  pack(P(58), 3, 576, 576);
  gemm(col, P(59), nullptr, actA, 32768, 3, 576, 16, 0);
  final_tanh_nchw<<<gcount(2ull * 3 * 128 * 128, 256), 256, 0, stream>>>(actA, (float*)d_out, 2, 128, 128, 16);
}
